// SchNet_8435315769379
// MI455X (gfx1250) — matrix operand order repaired and checked by reading, not compiled, not run on hardware
//
#include <hip/hip_runtime.h>
#include <hip/hip_bf16.h>

typedef __bf16 bf16_t;
typedef __bf16 v16bf __attribute__((ext_vector_type(16)));
typedef float  v8f   __attribute__((ext_vector_type(8)));

#define BATCH  2048
#define NELEC  16
#define NNUC   4
#define DIM    128
#define KDIM   64
#define FDIM   32
#define HDIM   45
#define E_SAME 112
#define E_ANTI 128
#define E_NE   64
#define E_TOT  304
#define CUT    10.0f

// ---------------- WMMA fragment helpers (CDNA5 16x16x32 bf16) ----------------
// A 16x32 (MxK): lanes 0-15 M=lane, lanes 16-31 M=lane-16;
// VGPR p<4: K = 2p + 8*half ; p>=4: K = 16 + 2(p-4) + 8*half  (pairs contiguous)
__device__ __forceinline__ v16bf load_a_frag(const bf16_t* A, int ld, int row0, int k0, int lane) {
  const int m = lane & 15, half = lane >> 4;
  const bf16_t* p = A + (size_t)(row0 + m) * ld + k0;
  v16bf a;
#pragma unroll
  for (int q = 0; q < 8; ++q) {
    const int k = (q < 4) ? (2 * q + 8 * half) : (16 + 2 * (q - 4) + 8 * half);
    a[2 * q]     = p[k];
    a[2 * q + 1] = p[k + 1];
  }
  return a;
}

// Pre-swizzle a KxN row-major bf16 matrix into per-fragment, per-lane contiguous
// LDS chunks following the B 32x16 layout (lanes 0-15 N=lane, K same as A fragment).
// Fragment f = kt*nn + ct occupies out[f*512 .. f*512+511]; lane L owns 16 bf16.
__device__ __forceinline__ void stage_b_frags(const bf16_t* Bm, int ld, int nk, int nn,
                                              bf16_t* out) {
  const int total = nk * nn * 32 * 8;            // one bf16 pair per index
  for (int i = threadIdx.x; i < total; i += (int)blockDim.x) {
    int q = i & 7, lane = (i >> 3) & 31, f = i >> 8;
    int kt = f / nn, ct = f - kt * nn;
    int n = lane & 15, half = lane >> 4;
    int k = kt * 32 + ((q < 4) ? (2 * q + 8 * half) : (16 + 2 * (q - 4) + 8 * half));
    int col = ct * 16 + n;
    out[i * 2]     = Bm[(size_t)k * ld + col];
    out[i * 2 + 1] = Bm[(size_t)(k + 1) * ld + col];
  }
}
__device__ __forceinline__ v16bf load_b_staged(const bf16_t* Bs, int f, int lane) {
  return *(const v16bf*)(Bs + ((size_t)f * 32 + lane) * 16);  // 32B aligned -> 2x ds_load_b128
}
__device__ __forceinline__ v8f wmma_bf16(v16bf a, v16bf b, v8f c) {
  return __builtin_amdgcn_wmma_f32_16x16x32_bf16(false, a, false, b, (short)0, c, false, false);
}

// ---------------- weight prep: fp32 -> padded bf16, inverse edge lists -------
__global__ void k_prep(const float* wW1, const float* wW2, const float* hW, const float* gW,
                       const int* same_r, const int* anti_r, const int* ne_r,
                       bf16_t* wW1b, bf16_t* wW2b, bf16_t* hWcat, bf16_t* gWcat,
                       int* inv_cnt, int* inv_edge) {
  const int tid = threadIdx.x;
  for (int i = tid; i < 3 * 3 * 32 * 48; i += blockDim.x) {        // [lj][32][48]
    int c = i % 48, r = (i / 48) % 32, lj = i / (48 * 32);
    wW1b[i] = (c < HDIM) ? (bf16_t)wW1[(lj * 32 + r) * HDIM + c] : (bf16_t)0.0f;
  }
  for (int i = tid; i < 3 * 3 * 64 * 64; i += blockDim.x) {        // [lj][64][64]
    int c = i % 64, r = (i / 64) % 64, lj = i / (64 * 64);
    wW2b[i] = (r < HDIM) ? (bf16_t)wW2[(lj * HDIM + r) * 64 + c] : (bf16_t)0.0f;
  }
  for (int i = tid; i < 2 * 128 * 128; i += blockDim.x) {          // hWcat [l][d][j*64+k]
    int c = i % 128, d = (i / 128) % 128, l = i / (128 * 128);
    int j = c >> 6, k = c & 63;
    hWcat[i] = (bf16_t)hW[(((l * 2 + j) * 128) + d) * 64 + k];
  }
  for (int i = tid; i < 3 * 192 * 128; i += blockDim.x) {          // gWcat [l][j*64+k][d]
    int d = i % 128, rk = (i / 128) % 192, l = i / (192 * 128);
    int j = rk >> 6, k = rk & 63;
    gWcat[i] = (bf16_t)gW[((l * 3 + j) * 64 + k) * 128 + d];
  }
  if (tid < 48) {  // receiver -> incoming-edge lists (<=8 each), deterministic scatter
    int j = tid >> 4, r = tid & 15;
    const int* recv = (j == 0) ? same_r : (j == 1) ? anti_r : ne_r;
    int E = (j == 0) ? E_SAME : (j == 1) ? E_ANTI : E_NE;
    int cnt = 0;
    for (int e = 0; e < E; ++e)
      if (recv[e] == r && cnt < 8) inv_edge[tid * 8 + cnt++] = e;
    inv_cnt[tid] = cnt;
  }
}

// ---------------- elec init: broadcast X_tab row ----------------
__global__ void k_init(const float* X_tab, float* out, bf16_t* elecb) {
  size_t i = (size_t)blockIdx.x * blockDim.x + threadIdx.x;
  if (i >= (size_t)BATCH * NELEC * DIM) return;
  float v = X_tab[i % DIM];
  out[i] = v;
  elecb[i] = (bf16_t)v;
}

// ---------------- distance featurization -> bf16 GEMM operand ----------------
__global__ void k_feats(const float* rs, const float* coords,
                        const int* same_s, const int* same_r,
                        const int* anti_s, const int* anti_r,
                        const int* ne_s, const int* ne_r,
                        bf16_t* feats) {
  int gid = blockIdx.x * blockDim.x + threadIdx.x;
  if (gid >= BATCH * E_TOT) return;
  int b = gid / E_TOT, ge = gid % E_TOT;
  float px, py, pz, qx, qy, qz;
  size_t rowOut;
  if (ge < E_SAME) {
    int e = ge;
    const float* p = rs + ((size_t)b * NELEC + same_s[e]) * 3;
    const float* q = rs + ((size_t)b * NELEC + same_r[e]) * 3;
    px = p[0]; py = p[1]; pz = p[2]; qx = q[0]; qy = q[1]; qz = q[2];
    rowOut = (size_t)b * E_SAME + e;
  } else if (ge < E_SAME + E_ANTI) {
    int e = ge - E_SAME;
    const float* p = rs + ((size_t)b * NELEC + anti_s[e]) * 3;
    const float* q = rs + ((size_t)b * NELEC + anti_r[e]) * 3;
    px = p[0]; py = p[1]; pz = p[2]; qx = q[0]; qy = q[1]; qz = q[2];
    rowOut = (size_t)BATCH * E_SAME + (size_t)b * E_ANTI + e;
  } else {
    int e = ge - E_SAME - E_ANTI;
    const float* p = coords + (size_t)ne_s[e] * 3;
    const float* q = rs + ((size_t)b * NELEC + ne_r[e]) * 3;
    px = p[0]; py = p[1]; pz = p[2]; qx = q[0]; qy = q[1]; qz = q[2];
    rowOut = (size_t)BATCH * (E_SAME + E_ANTI) + (size_t)b * E_NE + e;
  }
  float dx = px - qx, dy = py - qy, dz = pz - qz;
  float d = sqrtf(dx * dx + dy * dy + dz * dz);
  float env = d * d * __expf(-d);
  bf16_t* o = feats + rowOut * FDIM;
#pragma unroll
  for (int f = 0; f < FDIM; ++f) {
    float q01 = (float)f * (1.0f / 31.0f);
    float mu = CUT * q01 * q01;
    float sg = (1.0f + CUT * q01) * (1.0f / 7.0f);
    float t = (d - mu) / sg;
    o[f] = (bf16_t)(env * __expf(-t * t));
  }
}

// ---------------- fused edge MLP: GEMM1 -> silu -> LDS -> GEMM2 -> *hx -------
// MODE 0: hx = h0_tab[j][0][col]             (l==0, e-e edges)
// MODE 1: hx = helec[b*16+sender][j*64+col]  (l>0,  e-e edges)
// MODE 2: hx = Y_w[ne_s[e]][col]             (e-n edges)
template <int MODE>
__global__ void k_edge_mlp(const bf16_t* __restrict__ feats, int M, int E,
                           const bf16_t* __restrict__ w1, const float* __restrict__ b1,
                           const bf16_t* __restrict__ w2,
                           const int* __restrict__ senders, const float* __restrict__ hxsrc,
                           int hx_ofs, float* __restrict__ weh) {
  __shared__ bf16_t w1s[3 * 512];     // 3 staged fragments   (32x48)
  __shared__ bf16_t w2s[8 * 512];     // 2x4 staged fragments (64x64)
  __shared__ bf16_t hid[8][16 * 64];  // per-wave hidden tile
  stage_b_frags(w1, 48, 1, 3, w1s);
  stage_b_frags(w2, 64, 2, 4, w2s);
  __syncthreads();
  const int wave = threadIdx.x >> 5, lane = threadIdx.x & 31;
  const int n = lane & 15, half = lane >> 4;
  // loop-invariant scalars -> registers before the loop
  float biasv[3];
#pragma unroll
  for (int ct = 0; ct < 3; ++ct) {
    int col = ct * 16 + n;
    biasv[ct] = (col < HDIM) ? b1[col] : 0.0f;
  }
  float h0v[4] = {0, 0, 0, 0};
  if (MODE == 0) {
#pragma unroll
    for (int ct = 0; ct < 4; ++ct) h0v[ct] = hxsrc[hx_ofs + ct * 16 + n];
  }
  const int ntiles = M >> 4;
  for (int t0 = blockIdx.x * 8; t0 < ntiles; t0 += gridDim.x * 8) {
    const int t = t0 + wave;
    const bool act = t < ntiles;
    if (act) {
      v16bf a = load_a_frag(feats, FDIM, t * 16, 0, lane);  // K=32 exact
#pragma unroll
      for (int ct = 0; ct < 3; ++ct) {                      // N=48
        v16bf bb = load_b_staged(w1s, ct, lane);
        v8f c = {0, 0, 0, 0, 0, 0, 0, 0};
        c = wmma_bf16(a, bb, c);
#pragma unroll
        for (int v = 0; v < 8; ++v) {
          int r = v + 8 * half, col = ct * 16 + n;
          float x = c[v] + biasv[ct];
          float s = x / (1.0f + __expf(-x));                // silu
          hid[wave][r * 64 + col] = (bf16_t)s;
        }
      }
#pragma unroll
      for (int v = 0; v < 8; ++v)                           // zero K-pad cols 48..63
        hid[wave][(v + 8 * half) * 64 + 48 + n] = (bf16_t)0.0f;
    }
    __syncthreads();
    if (act) {
      v16bf a0 = load_a_frag(hid[wave], 64, 0, 0, lane);
      v16bf a1 = load_a_frag(hid[wave], 64, 0, 32, lane);
#pragma unroll
      for (int ct = 0; ct < 4; ++ct) {                      // N=64, K=64
        v8f c = {0, 0, 0, 0, 0, 0, 0, 0};
        c = wmma_bf16(a0, load_b_staged(w2s, ct, lane), c);
        c = wmma_bf16(a1, load_b_staged(w2s, 4 + ct, lane), c);
#pragma unroll
        for (int v = 0; v < 8; ++v) {
          int row = t * 16 + v + 8 * half;
          int col = ct * 16 + n;
          int b = row / E, e = row - b * E;
          float hx;
          if (MODE == 0)      hx = h0v[ct];
          else if (MODE == 1) hx = hxsrc[(size_t)(b * NELEC + senders[e]) * DIM + hx_ofs + col];
          else                hx = hxsrc[(size_t)senders[e] * KDIM + col];
          weh[(size_t)row * KDIM + col] = c[v] * hx;
        }
      }
    }
    __syncthreads();
  }
}

// ---------------- deterministic receiver aggregation -> zcat (bf16) ----------
__global__ void k_aggregate(const float* __restrict__ weh, int E, int j,
                            const int* __restrict__ inv_cnt, const int* __restrict__ inv_edge,
                            bf16_t* __restrict__ zcat) {
  int gid = blockIdx.x * blockDim.x + threadIdx.x;
  if (gid >= BATCH * NELEC * KDIM) return;
  int k = gid & 63, r = (gid >> 6) & 15, b = gid >> 10;
  int t = j * 16 + r;
  int cnt = inv_cnt[t];
  float s = 0.0f;
  for (int i = 0; i < cnt; ++i) {
    int e = inv_edge[t * 8 + i];
    s += weh[((size_t)b * E + e) * KDIM + k];
  }
  zcat[(size_t)(b * NELEC + r) * 192 + j * 64 + k] = (bf16_t)s;
}

// ---------------- helec = elec @ [hW_l0 | hW_l1]  (32768 x 128 x 128) --------
// One wave per 16-row tile; acc = 64 VGPRs; K loop kept rolled to bound pressure.
__global__ void k_hgemm(const bf16_t* __restrict__ elecb, const bf16_t* __restrict__ hWcatl,
                        float* __restrict__ helec) {
  __shared__ bf16_t Bs[4 * 8 * 512];  // staged fragments, 32KB
  stage_b_frags(hWcatl, DIM, 4, 8, Bs);
  __syncthreads();
  const int lane = threadIdx.x & 31;
  const int n = lane & 15, half = lane >> 4;
  int gw = (blockIdx.x * blockDim.x + threadIdx.x) >> 5;
  int nw = (gridDim.x * blockDim.x) >> 5;
  const int ntiles = (BATCH * NELEC) >> 4;
  for (int t = gw; t < ntiles; t += nw) {
    asm volatile("" ::: "memory");                         // keep LDS B-frag loads in-loop
    const int row0 = t * 16;
    v8f acc[8];
#pragma unroll
    for (int ct = 0; ct < 8; ++ct) acc[ct] = (v8f){0, 0, 0, 0, 0, 0, 0, 0};
#pragma unroll 1
    for (int kt = 0; kt < 4; ++kt) {
      v16bf a = load_a_frag(elecb, DIM, row0, kt * 32, lane);
#pragma unroll
      for (int ct = 0; ct < 8; ++ct)
        acc[ct] = wmma_bf16(a, load_b_staged(Bs, kt * 8 + ct, lane), acc[ct]);
    }
#pragma unroll
    for (int ct = 0; ct < 8; ++ct)
#pragma unroll
      for (int v = 0; v < 8; ++v)
        helec[(size_t)(row0 + v + 8 * half) * DIM + ct * 16 + n] = acc[ct][v];
  }
}

// ---------------- elec += zcat @ gWcat_l  (32768 x 192 x 128) ----------------
__global__ void k_ggemm_add(const bf16_t* __restrict__ zcat, const bf16_t* __restrict__ gWcatl,
                            float* __restrict__ out, bf16_t* __restrict__ elecb) {
  __shared__ bf16_t Bs[6 * 8 * 512];  // staged fragments, 48KB
  stage_b_frags(gWcatl, DIM, 6, 8, Bs);
  __syncthreads();
  const int lane = threadIdx.x & 31;
  const int n = lane & 15, half = lane >> 4;
  int gw = (blockIdx.x * blockDim.x + threadIdx.x) >> 5;
  int nw = (gridDim.x * blockDim.x) >> 5;
  const int ntiles = (BATCH * NELEC) >> 4;
  for (int t = gw; t < ntiles; t += nw) {
    asm volatile("" ::: "memory");
    const int row0 = t * 16;
    v8f acc[8];
#pragma unroll
    for (int ct = 0; ct < 8; ++ct) acc[ct] = (v8f){0, 0, 0, 0, 0, 0, 0, 0};
#pragma unroll 1
    for (int kt = 0; kt < 6; ++kt) {
      v16bf a = load_a_frag(zcat, 192, row0, kt * 32, lane);
#pragma unroll
      for (int ct = 0; ct < 8; ++ct)
        acc[ct] = wmma_bf16(a, load_b_staged(Bs, kt * 8 + ct, lane), acc[ct]);
    }
#pragma unroll
    for (int ct = 0; ct < 8; ++ct)
#pragma unroll
      for (int v = 0; v < 8; ++v) {
        size_t idx = (size_t)(row0 + v + 8 * half) * DIM + ct * 16 + n;
        float val = out[idx] + acc[ct][v];
        out[idx] = val;
        elecb[idx] = (bf16_t)val;
      }
  }
}

// ---------------- host orchestration ----------------
extern "C" void kernel_launch(void* const* d_in, const int* in_sizes, int n_in,
                              void* d_out, int out_size, void* d_ws, size_t ws_size,
                              hipStream_t stream) {
  (void)in_sizes; (void)n_in; (void)out_size; (void)ws_size;
  const float* rs     = (const float*)d_in[0];
  const float* coords = (const float*)d_in[1];
  const float* X_tab  = (const float*)d_in[2];
  const float* Y_w    = (const float*)d_in[3];
  const float* wW1    = (const float*)d_in[4];
  const float* wb1    = (const float*)d_in[5];
  const float* wW2    = (const float*)d_in[6];
  const float* h0_tab = (const float*)d_in[7];
  const float* hW     = (const float*)d_in[8];
  const float* gW     = (const float*)d_in[9];
  const int* same_s = (const int*)d_in[10];
  const int* same_r = (const int*)d_in[11];
  const int* anti_s = (const int*)d_in[12];
  const int* anti_r = (const int*)d_in[13];
  const int* ne_s   = (const int*)d_in[14];
  const int* ne_r   = (const int*)d_in[15];
  float* out = (float*)d_out;

  char* w = (char*)d_ws;
  size_t o = 0;
  auto take = [&](size_t bytes) -> void* {
    void* p = w + o;
    o = (o + bytes + 255) & ~(size_t)255;
    return p;
  };
  bf16_t* wW1b  = (bf16_t*)take((size_t)3 * 3 * 32 * 48 * 2);
  bf16_t* wW2b  = (bf16_t*)take((size_t)3 * 3 * 64 * 64 * 2);
  bf16_t* hWcat = (bf16_t*)take((size_t)2 * 128 * 128 * 2);
  bf16_t* gWcat = (bf16_t*)take((size_t)3 * 192 * 128 * 2);
  int* inv_cnt  = (int*)take(48 * 4);
  int* inv_edge = (int*)take(48 * 8 * 4);
  bf16_t* feats = (bf16_t*)take((size_t)BATCH * E_TOT * FDIM * 2);
  float* weh    = (float*)take((size_t)BATCH * E_ANTI * KDIM * 4);   // max E
  bf16_t* zcat  = (bf16_t*)take((size_t)BATCH * NELEC * 192 * 2);
  float* helec  = (float*)take((size_t)BATCH * NELEC * DIM * 4);
  bf16_t* elecb = (bf16_t*)take((size_t)BATCH * NELEC * DIM * 2);

  k_prep<<<1, 256, 0, stream>>>(wW1, wW2, hW, gW, same_r, anti_r, ne_r,
                                wW1b, wW2b, hWcat, gWcat, inv_cnt, inv_edge);
  {
    size_t tot = (size_t)BATCH * NELEC * DIM;
    k_init<<<(unsigned)((tot + 255) / 256), 256, 0, stream>>>(X_tab, out, elecb);
  }
  {
    int tot = BATCH * E_TOT;
    k_feats<<<(tot + 255) / 256, 256, 0, stream>>>(rs, coords, same_s, same_r,
                                                   anti_s, anti_r, ne_s, ne_r, feats);
  }

  const bf16_t* featsJ[3] = { feats,
                              feats + (size_t)BATCH * E_SAME * FDIM,
                              feats + (size_t)BATCH * (E_SAME + E_ANTI) * FDIM };
  const int EJ[3] = { E_SAME, E_ANTI, E_NE };
  const int* sendJ[3] = { same_s, anti_s, ne_s };

  for (int l = 0; l < 3; ++l) {
    if (l > 0)
      k_hgemm<<<256, 256, 0, stream>>>(elecb, hWcat + (size_t)(l - 1) * 128 * 128, helec);
    for (int j = 0; j < 3; ++j) {
      int M = BATCH * EJ[j];
      int ntiles = M / 16;
      int blocks = (ntiles + 7) / 8; if (blocks > 2048) blocks = 2048;
      const bf16_t* w1 = wW1b + (size_t)(l * 3 + j) * 32 * 48;
      const float*  b1 = wb1 + (size_t)(l * 3 + j) * HDIM;
      const bf16_t* w2 = wW2b + (size_t)(l * 3 + j) * 64 * 64;
      if (j == 2)
        k_edge_mlp<2><<<blocks, 256, 0, stream>>>(featsJ[j], M, EJ[j], w1, b1, w2,
                                                  sendJ[j], Y_w, 0, weh);
      else if (l == 0)
        k_edge_mlp<0><<<blocks, 256, 0, stream>>>(featsJ[j], M, EJ[j], w1, b1, w2,
                                                  sendJ[j], h0_tab, j * 64, weh);
      else
        k_edge_mlp<1><<<blocks, 256, 0, stream>>>(featsJ[j], M, EJ[j], w1, b1, w2,
                                                  sendJ[j], helec, j * 64, weh);
      {
        size_t tot = (size_t)BATCH * NELEC * KDIM;
        k_aggregate<<<(unsigned)((tot + 255) / 256), 256, 0, stream>>>(weh, EJ[j], j,
                                                                       inv_cnt, inv_edge, zcat);
      }
    }
    k_ggemm_add<<<256, 256, 0, stream>>>(zcat, gWcat + (size_t)l * 192 * 128, out, elecb);
  }
}
